// PAM_Module_69707319214289
// MI455X (gfx1250) — matrix operand order repaired and checked by reading, not compiled, not run on hardware
//
#include <hip/hip_runtime.h>
#include <cstdint>
#include <cstddef>

// ---------------------------------------------------------------------------
// PAM (position attention) fused flash-attention for MI455X (gfx1250).
// B=4, C=256, Cq=128, N=4096.  All matmuls on v_wmma_f32_16x16x32_bf16.
// Workspace layout (assumes ws_size >= ~17 MB):
//   [0x000000) Wall bf16 [512][256]   (rows 0..127 = Wq/16, 128..255 = Wk, 256..511 = Wv)
//   [0x040000) ball f32  [512]
//   [0x050000) qT  bf16  [B][N][128]  (token-major, pre-scaled by 1/sqrt(C))
//   [0x450000) kT  bf16  [B][N][128]  (token-major)
//   [0x850000) vbf bf16  [B][256][N]  (channel-major)
// ---------------------------------------------------------------------------

typedef __attribute__((ext_vector_type(16))) __bf16 v16bf;
typedef __attribute__((ext_vector_type(8)))  __bf16 v8bf;
typedef __attribute__((ext_vector_type(8)))  float  v8f;

#define CC   256
#define CQ   128
#define NN   4096
#define BB   4

#define WALL_OFF 0x000000
#define BALL_OFF 0x040000
#define QT_OFF   0x050000
#define KT_OFF   0x450000
#define VB_OFF   0x850000

__device__ __forceinline__ v8f wmma_bf16(v16bf a, v16bf b, v8f c) {
  // (neg_a, A, neg_b, B, c_mod, C, reuse_a, reuse_b)
  return __builtin_amdgcn_wmma_f32_16x16x32_bf16(false, a, false, b, (short)0, c,
                                                 false, false);
}

// -------------------------------- prep -------------------------------------
__global__ __launch_bounds__(256) void pam_prep(
    const float* __restrict__ Wq, const float* __restrict__ bq,
    const float* __restrict__ Wk, const float* __restrict__ bk,
    const float* __restrict__ Wv, const float* __restrict__ bv,
    __bf16* __restrict__ wall, float* __restrict__ ball) {
  int o = blockIdx.x;      // 0..511
  int c = threadIdx.x;     // 0..255
  float w;
  if (o < 128)      w = Wq[o * CC + c] * 0.0625f;   // fold 1/sqrt(256)
  else if (o < 256) w = Wk[(o - 128) * CC + c];
  else              w = Wv[(o - 256) * CC + c];
  wall[o * CC + c] = (__bf16)w;
  if (c == 0) {
    float b = (o < 128) ? bq[o] * 0.0625f : (o < 256) ? bk[o - 128] : bv[o - 256];
    ball[o] = b;
  }
}

// ------------------------------ projection ---------------------------------
// grid (N/64, B), 256 thr. LDS holds x^T tile [64 tok][256 ch] bf16 (padded).
#define XROW 528   // 256*2 + 16 pad, 16B multiple, conflict-free A reads

__global__ __launch_bounds__(256) void pam_proj(
    const float* __restrict__ x, const __bf16* __restrict__ wall,
    const float* __restrict__ ball,
    __bf16* __restrict__ qT, __bf16* __restrict__ kT, __bf16* __restrict__ vbf) {
  __shared__ __align__(16) char xs[64 * XROW];
  const int b   = blockIdx.y;
  const int n0  = blockIdx.x * 64;
  const int tid = threadIdx.x;

  // stage x[b, :, n0:n0+64] -> LDS as bf16, token-major
  {
    int nloc = tid & 63, c0 = tid >> 6;
    const float* xb = x + (size_t)b * CC * NN + n0 + nloc;
    for (int c = c0; c < CC; c += 4) {
      float f = xb[(size_t)c * NN];
      *(__bf16*)(xs + nloc * XROW + c * 2) = (__bf16)f;
    }
  }
  __syncthreads();

  const int w = tid >> 5, l = tid & 31;
  const int lo = l & 15, khalf = l >> 4;
  const int nbase = (w & 3) * 16;      // 16-token group
  const int o0    = (w >> 2) * 256;    // output half

  v8f acc[16];
  for (int ot = 0; ot < 16; ++ot) {
    float bbv = ball[o0 + ot * 16 + lo];
    for (int r = 0; r < 8; ++r) acc[ot][r] = bbv;
  }

  for (int k4 = 0; k4 < 8; ++k4) {
    // A = x^T tile: row n = nbase+lo; elems e: c=k4*32+khalf*16+e
    const char* ar = xs + (nbase + lo) * XROW + k4 * 64 + khalf * 32;
    union { v16bf v; v8bf h[2]; } A;
    A.h[0] = *(const v8bf*)(ar);
    A.h[1] = *(const v8bf*)(ar + 16);
    for (int ot = 0; ot < 16; ++ot) {
      int o = o0 + ot * 16 + lo;
      // B = Wall^T: col o, c = k4*32 + khalf*16 + e (32 contiguous bytes)
      v16bf Bv = *(const v16bf*)(wall + (size_t)o * CC + k4 * 32 + khalf * 16);
      acc[ot] = wmma_bf16(A.v, Bv, acc[ot]);
    }
  }

  // scatter outputs: D layout lane: col o = lo, rows n = 8*khalf + r
  for (int ot = 0; ot < 16; ++ot) {
    int o = o0 + ot * 16 + lo;
    if (o < 128) {
      for (int r = 0; r < 8; ++r) {
        int n = n0 + nbase + khalf * 8 + r;
        qT[((size_t)b * NN + n) * CQ + o] = (__bf16)acc[ot][r];
      }
    } else if (o < 256) {
      for (int r = 0; r < 8; ++r) {
        int n = n0 + nbase + khalf * 8 + r;
        kT[((size_t)b * NN + n) * CQ + (o - 128)] = (__bf16)acc[ot][r];
      }
    } else {
      int n = n0 + nbase + khalf * 8;         // 8 consecutive tokens
      v8bf pk;
      for (int r = 0; r < 8; ++r) pk[r] = (__bf16)acc[ot][r];
      *(v8bf*)(vbf + ((size_t)b * CC + (o - 256)) * NN + n) = pk;
    }
  }
}

// --------------------------- flash attention -------------------------------
// grid (N/128, B), 256 thr = 8 waves x 16 queries. BK = 32 keys / iter.
#define SKROW 272                    // K tile row stride (128 bf16 + pad)
#define SVROW 80                     // V tile row stride (32 bf16 + pad)
#define KT_BYTES (32 * SKROW)        // 8704
#define VT_BYTES (CC * SVROW)        // 20480
#define BUF_BYTES (KT_BYTES + VT_BYTES)
#define NIT (NN / 32)

__global__ __launch_bounds__(256) void pam_attn(
    const float* __restrict__ x, const float* __restrict__ gamma,
    const __bf16* __restrict__ qT, const __bf16* __restrict__ kT,
    const __bf16* __restrict__ vbf, float* __restrict__ out) {
  __shared__ __align__(16) char smem[2 * BUF_BYTES];

  const int b   = blockIdx.y;
  const int tid = threadIdx.x;
  const int l = tid & 31, w = tid >> 5;
  const int lo = l & 15, khalf = l >> 4;
  const int q0 = blockIdx.x * 128 + w * 16;

  const __bf16* kTb = kT + (size_t)b * NN * CQ;
  const __bf16* vb  = vbf + (size_t)b * CC * NN;

  // async double-buffered staging of K (32x128) and V (256x32) bf16 tiles
  auto stage = [&](int buf, int it) {
    const int n0 = it * 32;
    uint32_t kbase = (uint32_t)(uintptr_t)(&smem[0]) + buf * BUF_BYTES;
    uint32_t vbase = kbase + KT_BYTES;
#pragma unroll
    for (int j = 0; j < 2; ++j) {                 // 512 x 16B chunks of K
      int idx = tid + j * 256;
      int row = idx >> 4, col = idx & 15;
      uint32_t lds  = kbase + row * SKROW + col * 16;
      uint32_t goff = ((uint32_t)(n0 + row) * CQ + col * 8) * 2;
      asm volatile("global_load_async_to_lds_b128 %0, %1, %2"
                   :: "v"(lds), "v"(goff), "s"(kTb) : "memory");
    }
#pragma unroll
    for (int j = 0; j < 4; ++j) {                 // 1024 x 16B chunks of V
      int idx = tid + j * 256;
      int row = idx >> 2, col = idx & 3;
      uint32_t lds  = vbase + row * SVROW + col * 16;
      uint32_t goff = ((uint32_t)row * NN + n0 + col * 8) * 2;
      asm volatile("global_load_async_to_lds_b128 %0, %1, %2"
                   :: "v"(lds), "v"(goff), "s"(vb) : "memory");
    }
  };

  // Q in B-layout (col m = lo, c = k4*32 + khalf*16 + e), loaded once
  v16bf Qb[4];
  {
    const __bf16* qrow = qT + ((size_t)b * NN + q0 + lo) * CQ + khalf * 16;
#pragma unroll
    for (int k4 = 0; k4 < 4; ++k4) Qb[k4] = *(const v16bf*)(qrow + k4 * 32);
  }

  v8f O[16];
  for (int t = 0; t < 16; ++t)
    for (int r = 0; r < 8; ++r) O[t][r] = 0.f;
  float mrun = -1e30f, lrun = 0.f;

  stage(0, 0);
  for (int it = 0; it < NIT; ++it) {
    asm volatile("s_wait_asynccnt 0x0" ::: "memory");
    __syncthreads();
    if (it + 1 < NIT) stage((it + 1) & 1, it + 1);

    const char* kb  = smem + (it & 1) * BUF_BYTES;
    const char* vbl = kb + KT_BYTES;

    // S^T = K^T * Q : two 16-key tiles, contraction Cq=128 in 4 chunks
    v8f S0, S1;
    for (int r = 0; r < 8; ++r) { S0[r] = 0.f; S1[r] = 0.f; }
#pragma unroll
    for (int k4 = 0; k4 < 4; ++k4) {
      const char* a0 = kb + lo * SKROW        + k4 * 64 + khalf * 32;
      const char* a1 = kb + (16 + lo) * SKROW + k4 * 64 + khalf * 32;
      union { v16bf v; v8bf h[2]; } A0, A1;
      A0.h[0] = *(const v8bf*)a0; A0.h[1] = *(const v8bf*)(a0 + 16);
      A1.h[0] = *(const v8bf*)a1; A1.h[1] = *(const v8bf*)(a1 + 16);
      S0 = wmma_bf16(A0.v, Qb[k4], S0);
      S1 = wmma_bf16(A1.v, Qb[k4], S1);
    }

    // online softmax: lane owns query m = lo; keys n = 8*khalf+r (S0), +16 (S1)
    float mt = -1e30f;
#pragma unroll
    for (int r = 0; r < 8; ++r) { mt = fmaxf(mt, S0[r]); mt = fmaxf(mt, S1[r]); }
    mt = fmaxf(mt, __shfl_xor(mt, 16));
    float mnew  = fmaxf(mrun, mt);
    float scale = __expf(mrun - mnew);

    union { v16bf v; __bf16 e[16]; } P;   // A-layout of P, elementwise = S^T D regs
    float lt = 0.f;
#pragma unroll
    for (int r = 0; r < 8; ++r) {
      float p0 = __expf(S0[r] - mnew);
      float p1 = __expf(S1[r] - mnew);
      lt += p0 + p1;
      P.e[r]     = (__bf16)p0;
      P.e[r + 8] = (__bf16)p1;
    }
    lt += __shfl_xor(lt, 16);
    lrun = lrun * scale + lt;
    mrun = mnew;

    // rescale O (D layout rows m = 8*khalf + r -> gather scale from lane m)
    float fac[8];
#pragma unroll
    for (int r = 0; r < 8; ++r) fac[r] = __shfl(scale, khalf * 8 + r);
#pragma unroll
    for (int t = 0; t < 16; ++t)
#pragma unroll
      for (int r = 0; r < 8; ++r) O[t][r] *= fac[r];

    // O += P * V : 16 channel tiles; V B-layout: col c, rows n = khalf*8 + e (e<8), 16 + khalf*8 + (e-8) (e>=8)
#pragma unroll
    for (int t = 0; t < 16; ++t) {
      const char* vp = vbl + (t * 16 + lo) * SVROW + khalf * 16;
      union { v16bf v; v8bf h[2]; } Bv;
      Bv.h[0] = *(const v8bf*)vp;
      Bv.h[1] = *(const v8bf*)(vp + 32);
      O[t] = wmma_bf16(P.v, Bv.v, O[t]);
    }
  }

  // epilogue: out = tanh(gamma * O/(lsum*N) + x)
  float inv = 1.0f / (lrun * (float)NN);
  float g   = gamma[0];
  float facn[8];
#pragma unroll
  for (int r = 0; r < 8; ++r) facn[r] = __shfl(inv, khalf * 8 + r);
  const float* xb = x + (size_t)b * CC * NN;
  float*       ob = out + (size_t)b * CC * NN;
  for (int t = 0; t < 16; ++t) {
    int c = t * 16 + lo;
#pragma unroll
    for (int r = 0; r < 8; ++r) {
      int m = q0 + khalf * 8 + r;
      size_t idx = (size_t)c * NN + m;
      float v = O[t][r] * facn[r];
      ob[idx] = tanhf(g * v + xb[idx]);
    }
  }
}

// ------------------------------- launcher ----------------------------------
extern "C" void kernel_launch(void* const* d_in, const int* in_sizes, int n_in,
                              void* d_out, int out_size, void* d_ws, size_t ws_size,
                              hipStream_t stream) {
  const float* x     = (const float*)d_in[0];
  const float* Wq    = (const float*)d_in[1];
  const float* bq    = (const float*)d_in[2];
  const float* Wk    = (const float*)d_in[3];
  const float* bk    = (const float*)d_in[4];
  const float* Wv    = (const float*)d_in[5];
  const float* bv    = (const float*)d_in[6];
  const float* gamma = (const float*)d_in[7];
  float* out = (float*)d_out;

  char* ws = (char*)d_ws;
  __bf16* wall = (__bf16*)(ws + WALL_OFF);
  float*  ball = (float*)(ws + BALL_OFF);
  __bf16* qT   = (__bf16*)(ws + QT_OFF);
  __bf16* kT   = (__bf16*)(ws + KT_OFF);
  __bf16* vbf  = (__bf16*)(ws + VB_OFF);

  pam_prep<<<512, 256, 0, stream>>>(Wq, bq, Wk, bk, Wv, bv, wall, ball);
  pam_proj<<<dim3(NN / 64, BB), 256, 0, stream>>>(x, wall, ball, qT, kT, vbf);
  pam_attn<<<dim3(NN / 128, BB), 256, 0, stream>>>(x, gamma, qT, kT, vbf, out);
}
